// MobaAttention_58841051955743
// MI455X (gfx1250) — matrix operand order repaired and checked by reading, not compiled, not run on hardware
//
#include <hip/hip_runtime.h>
#include <math.h>

#define HIDDEN  2048
#define NHEADS  32
#define HEADDIM 64
#define SEQ     2048
#define CHUNK   256
#define NCHUNK  8
#define TOPK    4

typedef __attribute__((ext_vector_type(16))) _Float16 v16h;
typedef __attribute__((ext_vector_type(8)))  _Float16 v8h;
typedef __attribute__((ext_vector_type(8)))  float    v8f;
typedef __attribute__((ext_vector_type(4)))  unsigned int u32x4;
typedef __attribute__((ext_vector_type(8)))  unsigned int u32x8;

// ---------------------------------------------------------------------------
// A-operand loader: 16-bit A-matrix 16x32 layout (ISA 7.12.2).
// lane l: row = l%16, half = l/16. VGPR0..3 hold K = half*8 + 0..7,
// VGPR4..7 hold K = 16 + half*8 + 0..7. Caller passes p already offset by
// (k_base + half*8); we read [p, p+8) and [p+16, p+24).
// ---------------------------------------------------------------------------
__device__ __forceinline__ v16h load_a_frag(const _Float16* p) {
  v8h lo = *reinterpret_cast<const v8h*>(p);
  v8h hi = *reinterpret_cast<const v8h*>(p + 16);
  v16h r;
#pragma unroll
  for (int i = 0; i < 8; ++i) { r[i] = lo[i]; r[i + 8] = hi[i]; }
  return r;
}

// B-operand loader: 16 contiguous halves starting at p (16B-aligned is enough)
__device__ __forceinline__ v16h load_b_frag(const _Float16* p) {
  v8h lo = *reinterpret_cast<const v8h*>(p);
  v8h hi = *reinterpret_cast<const v8h*>(p + 8);
  v16h r;
#pragma unroll
  for (int i = 0; i < 8; ++i) { r[i] = lo[i]; r[i + 8] = hi[i]; }
  return r;
}

__device__ __forceinline__ v8f wmma_f16(v16h a, v16h b, v8f c) {
  return __builtin_amdgcn_wmma_f32_16x16x32_f16(
      false, a, false, b, (short)0, c, false, false);
}

// Low 32 bits of a generic pointer to a __shared__ object = LDS-relative
// byte offset (ISA 10.2: LDS aperture truncates to addr[31:0]).
__device__ __forceinline__ unsigned lds_off_of(const void* p) {
  return (unsigned)(unsigned long long)p;
}

// Per-lane async global->LDS copy of 16 bytes (ASYNCcnt).
__device__ __forceinline__ void async_ld_b128(unsigned lds_off,
                                              const void* gaddr) {
  asm volatile("global_load_async_to_lds_b128 %0, %1, off"
               :: "v"(lds_off), "v"(gaddr) : "memory");
}
__device__ __forceinline__ void wait_async0() {
  asm volatile("s_wait_asynccnt 0" ::: "memory");
}

// ---------------------------------------------------------------------------
// Tensor Data Mover: 2D tile load Global->LDS (TENSORcnt).
// Loads `rows` rows of `tile_dw` dwords (row stride ld_dw dwords) starting at
// gsrc into LDS at ldsoff. pad_enable inserts 4 dwords of LDS padding after
// every 32 dwords stored -> LDS row stride = 36 dwords = 72 halves.
// D# layout per CDNA5 ISA (2-group form, <=2D).
// ---------------------------------------------------------------------------
__device__ __forceinline__ void tdm_load_2d(const void* gsrc, unsigned ldsoff,
                                            unsigned tile_dw, unsigned rows,
                                            unsigned ld_dw) {
  unsigned long long ga = (unsigned long long)gsrc;
  u32x4 g0;
  g0[0] = 1u;                                   // count=1, user descriptor
  g0[1] = ldsoff;                               // lds_addr [63:32]
  g0[2] = (unsigned)ga;                         // global_addr [95:64]
  g0[3] = (unsigned)((ga >> 32) & 0x01ffffffull)// global_addr [120:96]
          | (2u << 30);                         // type=2 ("image") [127:126]
  u32x8 g1;
  g1[0] = (2u << 16)                            // data_size = 4B
        | (1u << 20)                            // pad_enable
        | (4u << 22)                            // pad_interval: 32 dwords
        | (3u << 25);                           // pad_amount: 4 dwords
  g1[1] = (ld_dw & 0xffffu) << 16;              // tensor_dim0[15:0] @ [63:48]
  g1[2] = (ld_dw >> 16) | ((rows & 0xffffu) << 16); // dim0 hi | tensor_dim1 lo
  g1[3] = (rows >> 16) | (tile_dw << 16);       // dim1 hi | tile_dim0
  g1[4] = rows & 0xffffu;                       // tile_dim1 | tile_dim2=0
  g1[5] = ld_dw;                                // tensor_dim0_stride lo32
  g1[6] = 0;                                    // stride hi | dim1_stride lo
  g1[7] = 0;
  asm volatile("tensor_load_to_lds %0, %1" :: "s"(g0), "s"(g1) : "memory");
}

// ---------------------------------------------------------------------------
// Elementwise f32 -> f16
// ---------------------------------------------------------------------------
__global__ void cvt_f32_f16(const float* __restrict__ in,
                            _Float16* __restrict__ out, int n) {
  int i = blockIdx.x * blockDim.x + threadIdx.x;
  if (i < n) out[i] = (_Float16)in[i];
}

// ---------------------------------------------------------------------------
// C[M,N] (f32) = A[M,K] (f16, ld=lda) * B[N,K]^T (f16, ld=ldb)
// Block = 128 threads (4 waves), block tile 64M x 64N, K chunk = 64.
// B tile (shared by all 4 waves) staged by TDM tensor_load_to_lds;
// A tile staged by per-lane global_load_async_to_lds_b128.
// Double-buffered: next chunk's DMA overlaps current chunk's 8 WMMAs/wave.
// grid = (N/64, M/64).
// ---------------------------------------------------------------------------
__global__ void __launch_bounds__(128) gemm_nt_f16(
    const _Float16* __restrict__ A, int lda,
    const _Float16* __restrict__ B, int ldb,
    float* __restrict__ C, int ldc, int K) {
  __shared__ _Float16 lA[2][64][72];   // 72-half row stride (pad 8)
  __shared__ _Float16 lB[2][64][72];

  const int tid  = threadIdx.x;
  const int lane = tid & 31;
  const int wave = tid >> 5;
  const int r    = lane & 15;
  const int hlf  = lane >> 4;
  const int m0   = blockIdx.y * 64;
  const int n0   = blockIdx.x * 64;
  const int mw   = wave * 16;          // wave's M strip within block tile

  auto stage = [&](int nb, int kc2) {
    // A tile: 64 rows x 64 halves = 512 x 16B segments, 4 per thread
#pragma unroll
    for (int j = 0; j < 4; ++j) {
      int c = j * 128 + tid;
      int row = c >> 3, sg = c & 7;
      const _Float16* gp = A + (size_t)(m0 + row) * lda + kc2 + sg * 8;
      async_ld_b128(lds_off_of(&lA[nb][row][sg * 8]), gp);
    }
    // B tile: one TDM descriptor issued by wave 0 only
    if (wave == 0)
      tdm_load_2d(B + (size_t)n0 * ldb + kc2, lds_off_of(&lB[nb][0][0]),
                  32u, 64u, (unsigned)(ldb >> 1));
  };

  v8f acc[4] = {};
  stage(0, 0);
  int cur = 0;

  for (int kc = 0; kc < K; kc += 64) {
    wait_async0();                               // my A segments landed
    if (wave == 0) __builtin_amdgcn_s_wait_tensorcnt(0);  // B tile landed
    __syncthreads();                             // publish cur, retire nxt
    int nxt = cur ^ 1;
    if (kc + 64 < K) stage(nxt, kc + 64);        // overlap DMA with compute
#pragma unroll
    for (int ks = 0; ks < 64; ks += 32) {
      v16h a = load_a_frag(&lA[cur][mw + r][ks + hlf * 8]);
#pragma unroll
      for (int nt = 0; nt < 4; ++nt) {
        v16h b = load_a_frag(&lB[cur][nt * 16 + r][ks + hlf * 8]);
        acc[nt] = wmma_f16(a, b, acc[nt]);
      }
    }
    cur = nxt;
  }

  // C/D layout: lane holds column n = 16*nt + r, rows m = v + 8*half
#pragma unroll
  for (int v = 0; v < 8; ++v) {
    float* crow = C + (size_t)(m0 + mw + v + 8 * hlf) * ldc + n0 + r;
#pragma unroll
    for (int nt = 0; nt < 4; ++nt) crow[nt * 16] = acc[nt][v];
  }
}

// ---------------------------------------------------------------------------
// RoPE in-place on f32 (S, H, D) + emit f16 copy. One thread per (s,h,i<32).
// ---------------------------------------------------------------------------
__global__ void rope_cvt(float* __restrict__ buf, _Float16* __restrict__ out16,
                         int total) {
  int idx = blockIdx.x * blockDim.x + threadIdx.x;
  if (idx >= total) return;
  int s   = idx / (NHEADS * 32);
  int rem = idx % (NHEADS * 32);
  int h   = rem >> 5;
  int i   = rem & 31;
  size_t base = (size_t)s * HIDDEN + h * HEADDIM;
  float t1 = buf[base + i], t2 = buf[base + 32 + i];
  float freq = (float)s * expf(-(float)i * (9.210340371976184f / 32.0f));
  float c = cosf(freq), sn = sinf(freq);
  float o1 = t1 * c - t2 * sn;
  float o2 = t1 * sn + t2 * c;
  buf[base + i]        = o1;
  buf[base + 32 + i]   = o2;
  out16[base + i]      = (_Float16)o1;
  out16[base + 32 + i] = (_Float16)o2;
}

// ---------------------------------------------------------------------------
// v (S, H*D) f32 -> vT (H*D, S) f16
// ---------------------------------------------------------------------------
__global__ void vtrans(const float* __restrict__ v, _Float16* __restrict__ vt,
                       int total) {
  int idx = blockIdx.x * blockDim.x + threadIdx.x;
  if (idx >= total) return;
  int s = idx / HIDDEN, c = idx % HIDDEN;
  vt[(size_t)c * SEQ + s] = (_Float16)v[idx];
}

// ---------------------------------------------------------------------------
// kmean[h][n][d] = mean over 256 positions of rope'd k. One thread per (n,h,d).
// ---------------------------------------------------------------------------
__global__ void kmean_kernel(const float* __restrict__ k,
                             float* __restrict__ km) {
  int idx = blockIdx.x * blockDim.x + threadIdx.x;
  if (idx >= NHEADS * NCHUNK * HEADDIM) return;
  int n  = idx / (NHEADS * HEADDIM);
  int rr = idx % (NHEADS * HEADDIM);   // rr = h*64 + d
  float sum = 0.f;
  const float* p = k + (size_t)n * CHUNK * HIDDEN + rr;
  for (int c = 0; c < CHUNK; ++c) sum += p[(size_t)c * HIDDEN];
  km[(size_t)(rr >> 6) * (NCHUNK * HEADDIM) + n * HEADDIM + (rr & 63)] =
      sum * (1.0f / CHUNK);
}

// ---------------------------------------------------------------------------
// gate = q . kmean, mask future (-1e30) / self (+1e30), top-4 -> 8-bit mask.
// One thread per (h, s).
// ---------------------------------------------------------------------------
__global__ void gate_topk(const float* __restrict__ q,
                          const float* __restrict__ km,
                          unsigned char* __restrict__ maskbuf) {
  int idx = blockIdx.x * blockDim.x + threadIdx.x;
  if (idx >= NHEADS * SEQ) return;
  int h = idx / SEQ, s = idx % SEQ;
  int qc = s >> 8;
  const float* qp = q + (size_t)s * HIDDEN + h * HEADDIM;
  const float* kp = km + (size_t)h * (NCHUNK * HEADDIM);
  float g[NCHUNK];
#pragma unroll
  for (int n = 0; n < NCHUNK; ++n) {
    if (n > qc)  { g[n] = -1e30f; continue; }
    if (n == qc) { g[n] =  1e30f; continue; }
    float d = 0.f;
    for (int i = 0; i < HEADDIM; ++i) d += qp[i] * kp[n * HEADDIM + i];
    g[n] = d;
  }
  unsigned m = 0;
#pragma unroll
  for (int t = 0; t < TOPK; ++t) {
    int bj = 0; float bv = -3.4e38f;
#pragma unroll
    for (int n = 0; n < NCHUNK; ++n)
      if (!((m >> n) & 1u) && g[n] > bv) { bv = g[n]; bj = n; }
    m |= 1u << bj;
  }
  maskbuf[(size_t)h * SEQ + s] = (unsigned char)m;
}

// ---------------------------------------------------------------------------
// MoBA attention. One wave per (head, 16-query tile). block=128 (4 waves),
// grid = (NHEADS, SEQ/64). Scores computed transposed (S^T = K * Q^T) so each
// lane owns one query column; softmax stats reduce via one xor-16 shuffle.
// O^T = V^T * P^T accumulates via WMMA.
// ---------------------------------------------------------------------------
__global__ void __launch_bounds__(128) moba_attn(
    const _Float16* __restrict__ q16, const _Float16* __restrict__ k16,
    const _Float16* __restrict__ vt,  const unsigned char* __restrict__ maskbuf,
    const float* __restrict__ gbuf,   const float* __restrict__ wnorm,
    _Float16* __restrict__ o16) {
  const int lane = threadIdx.x & 31;
  const int wave = threadIdx.x >> 5;
  const int r    = lane & 15;
  const int hlf  = lane >> 4;
  const int h    = blockIdx.x;
  const int q0   = (blockIdx.y * 4 + wave) * 16;
  const int qc   = q0 >> 8;
  const int sg   = q0 + r;
  const unsigned rm = maskbuf[(size_t)h * SEQ + sg];

  const _Float16* qrow = q16 + (size_t)sg * HIDDEN + h * HEADDIM + hlf * 8;
  v16h qf0 = load_a_frag(qrow);        // d = 0..31
  v16h qf1 = load_a_frag(qrow + 32);   // d = 32..63

  v8f acc[4] = {};
  float mrun = -1e30f, lrun = 0.f;

  for (int n = 0; n <= qc; ++n) {
    if (__ballot((int)((rm >> n) & 1u)) == 0) continue;   // wave-uniform skip
    const bool cok = (rm >> n) & 1u;
    const int tstart = n * CHUNK;
    const int tend   = (n == qc) ? (q0 + 16) : (n + 1) * CHUNK;

    for (int t0 = tstart; t0 < tend; t0 += 32) {
      float sA[8], sB[8];
      {
        const _Float16* kr =
            k16 + (size_t)(t0 + r) * HIDDEN + h * HEADDIM + hlf * 8;
        v8f st = {};
        st = wmma_f16(load_a_frag(kr),      qf0, st);
        st = wmma_f16(load_a_frag(kr + 32), qf1, st);
#pragma unroll
        for (int v = 0; v < 8; ++v) sA[v] = st[v];
      }
      {
        const _Float16* kr =
            k16 + (size_t)(t0 + 16 + r) * HIDDEN + h * HEADDIM + hlf * 8;
        v8f st = {};
        st = wmma_f16(load_a_frag(kr),      qf0, st);
        st = wmma_f16(load_a_frag(kr + 32), qf1, st);
#pragma unroll
        for (int v = 0; v < 8; ++v) sB[v] = st[v];
      }
#pragma unroll
      for (int v = 0; v < 8; ++v) {
        int tA = t0 + v + 8 * hlf;
        int tB = tA + 16;
        sA[v] = (cok && tA <= sg) ? sA[v] * 0.125f : -3.0e38f;
        sB[v] = (cok && tB <= sg) ? sB[v] * 0.125f : -3.0e38f;
      }
      float mloc = -3.0e38f;
#pragma unroll
      for (int v = 0; v < 8; ++v) mloc = fmaxf(mloc, fmaxf(sA[v], sB[v]));
      mloc = fmaxf(mloc, __shfl_xor(mloc, 16, 32));
      float mnew  = fmaxf(mrun, mloc);
      float alpha = expf(mrun - mnew);
      float pA[8], pB[8], ls = 0.f;
#pragma unroll
      for (int v = 0; v < 8; ++v) {
        pA[v] = expf(sA[v] - mnew);
        pB[v] = expf(sB[v] - mnew);
        ls += pA[v] + pB[v];
      }
      ls += __shfl_xor(ls, 16, 32);
      lrun = lrun * alpha + ls;
      mrun = mnew;
#pragma unroll
      for (int dt = 0; dt < 4; ++dt)
#pragma unroll
        for (int v = 0; v < 8; ++v) acc[dt][v] *= alpha;

      // P^T as B-operand: lanes 0-15 take t0+0..15, lanes 16-31 t0+16..31;
      // cross-half rows fetched with xor-16 shuffles.
      v16h pf;
#pragma unroll
      for (int v = 0; v < 8; ++v) {
        float xA = __shfl_xor(pA[v], 16, 32);
        float xB = __shfl_xor(pB[v], 16, 32);
        pf[v]     = (_Float16)(hlf ? xB    : pA[v]);
        pf[v + 8] = (_Float16)(hlf ? pB[v] : xA);
      }
#pragma unroll
      for (int dt = 0; dt < 4; ++dt) {
        const _Float16* vr =
            vt + (size_t)(h * HEADDIM + dt * 16 + r) * SEQ + t0 + hlf * 16;
        acc[dt] = wmma_f16(load_b_frag(vr), pf, acc[dt]);
      }
    }
  }

  float inv = 1.0f / lrun;
  float ss = 0.f;
#pragma unroll
  for (int dt = 0; dt < 4; ++dt)
#pragma unroll
    for (int v = 0; v < 8; ++v) {
      float o = acc[dt][v] * inv;
      acc[dt][v] = o;
      ss += o * o;
    }
  ss += __shfl_xor(ss, 16, 32);
  float rinv = rsqrtf(ss * (1.0f / HEADDIM) + 1e-6f);
#pragma unroll
  for (int dt = 0; dt < 4; ++dt)
#pragma unroll
    for (int v = 0; v < 8; ++v) {
      int d = dt * 16 + v + 8 * hlf;
      float gv  = gbuf[(size_t)sg * HIDDEN + h * HEADDIM + d];
      float sig = 1.0f / (1.0f + expf(-gv));
      float val = acc[dt][v] * rinv * wnorm[d] * sig;
      o16[(size_t)sg * HIDDEN + h * HEADDIM + d] = (_Float16)val;
    }
}

// ---------------------------------------------------------------------------
// host launcher
// ---------------------------------------------------------------------------
extern "C" void kernel_launch(void* const* d_in, const int* in_sizes, int n_in,
                              void* d_out, int out_size, void* d_ws,
                              size_t ws_size, hipStream_t stream) {
  const float* hsrc  = (const float*)d_in[0];
  const float* Wq    = (const float*)d_in[1];
  const float* Wk    = (const float*)d_in[2];
  const float* Wv    = (const float*)d_in[3];
  const float* Wo    = (const float*)d_in[4];
  const float* Wg1   = (const float*)d_in[5];
  const float* Wg2   = (const float*)d_in[6];
  const float* wnorm = (const float*)d_in[7];
  float* out = (float*)d_out;

  char* ws = (char*)d_ws;
  size_t off = 0;
  auto alloc = [&](size_t bytes) -> void* {
    void* p = ws + off;
    off += (bytes + 255) & ~(size_t)255;
    return p;
  };

  const size_t SH = (size_t)SEQ * HIDDEN;
  _Float16* h16  = (_Float16*)alloc(SH * 2);
  _Float16* wq16 = (_Float16*)alloc(SH * 2);
  _Float16* wk16 = (_Float16*)alloc(SH * 2);
  _Float16* wv16 = (_Float16*)alloc(SH * 2);
  _Float16* wo16 = (_Float16*)alloc(SH * 2);
  _Float16* wg1h = (_Float16*)alloc((size_t)HEADDIM * HIDDEN * 2);
  _Float16* wg2h = (_Float16*)alloc((size_t)HIDDEN * HEADDIM * 2);
  float*    qf   = (float*)alloc(SH * 4);
  float*    kf   = (float*)alloc(SH * 4);
  float*    vf   = (float*)alloc(SH * 4);
  _Float16* q16  = (_Float16*)alloc(SH * 2);
  _Float16* k16  = (_Float16*)alloc(SH * 2);
  _Float16* vt   = (_Float16*)alloc(SH * 2);
  float*    km   = (float*)alloc((size_t)NHEADS * NCHUNK * HEADDIM * 4);
  unsigned char* maskb = (unsigned char*)alloc((size_t)NHEADS * SEQ);
  float*    g1f  = (float*)alloc((size_t)SEQ * HEADDIM * 4);
  _Float16* g1h  = (_Float16*)alloc((size_t)SEQ * HEADDIM * 2);
  float*    gf   = (float*)alloc(SH * 4);
  _Float16* o16  = (_Float16*)alloc(SH * 2);

  const int NE  = SEQ * HIDDEN;
  const int NEg = HEADDIM * HIDDEN;
  const int T = 256;

  cvt_f32_f16<<<(NE + T - 1) / T, T, 0, stream>>>(hsrc, h16, NE);
  cvt_f32_f16<<<(NE + T - 1) / T, T, 0, stream>>>(Wq,  wq16, NE);
  cvt_f32_f16<<<(NE + T - 1) / T, T, 0, stream>>>(Wk,  wk16, NE);
  cvt_f32_f16<<<(NE + T - 1) / T, T, 0, stream>>>(Wv,  wv16, NE);
  cvt_f32_f16<<<(NE + T - 1) / T, T, 0, stream>>>(Wo,  wo16, NE);
  cvt_f32_f16<<<(NEg + T - 1) / T, T, 0, stream>>>(Wg1, wg1h, NEg);
  cvt_f32_f16<<<(NEg + T - 1) / T, T, 0, stream>>>(Wg2, wg2h, NEg);

  dim3 gqkv(HIDDEN / 64, SEQ / 64);
  gemm_nt_f16<<<gqkv, 128, 0, stream>>>(h16, HIDDEN, wq16, HIDDEN, qf, HIDDEN, HIDDEN);
  gemm_nt_f16<<<gqkv, 128, 0, stream>>>(h16, HIDDEN, wk16, HIDDEN, kf, HIDDEN, HIDDEN);
  gemm_nt_f16<<<gqkv, 128, 0, stream>>>(h16, HIDDEN, wv16, HIDDEN, vf, HIDDEN, HIDDEN);

  gemm_nt_f16<<<dim3(1, SEQ / 64), 128, 0, stream>>>(h16, HIDDEN, wg1h, HIDDEN,
                                                     g1f, HEADDIM, HIDDEN);
  {
    int n = SEQ * HEADDIM;
    cvt_f32_f16<<<(n + T - 1) / T, T, 0, stream>>>(g1f, g1h, n);
  }
  gemm_nt_f16<<<gqkv, 128, 0, stream>>>(g1h, HEADDIM, wg2h, HEADDIM, gf, HIDDEN,
                                        HEADDIM);

  {
    int n = SEQ * NHEADS * 32;
    rope_cvt<<<(n + T - 1) / T, T, 0, stream>>>(qf, q16, n);
    rope_cvt<<<(n + T - 1) / T, T, 0, stream>>>(kf, k16, n);
  }
  vtrans<<<(NE + T - 1) / T, T, 0, stream>>>(vf, vt, NE);

  {
    int n = NHEADS * NCHUNK * HEADDIM;
    kmean_kernel<<<(n + T - 1) / T, T, 0, stream>>>(kf, km);
    int m = NHEADS * SEQ;
    gate_topk<<<(m + T - 1) / T, T, 0, stream>>>(qf, km, maskb);
  }

  moba_attn<<<dim3(NHEADS, SEQ / 64), 128, 0, stream>>>(q16, k16, vt, maskb, gf,
                                                        wnorm, o16);

  gemm_nt_f16<<<gqkv, 128, 0, stream>>>(o16, HIDDEN, wo16, HIDDEN, out, HIDDEN,
                                        HIDDEN);
}
